// GATConv_52046413693117
// MI455X (gfx1250) — matrix operand order repaired and checked by reading, not compiled, not run on hardware
//
#include <hip/hip_runtime.h>
#include <hip/hip_bf16.h>

// ---------------- problem constants ----------------
#define N_NODES  100000
#define N_EDGES  800000
#define IN_CH    128
#define HC       128      // HEADS * OUT_CH
#define HEADS    4
#define OUT_CH   32
#define NPAD     100096   // N_NODES rounded up to 128
#define LDW      136      // padded bf16 LDS row stride (272B: 16B-aligned, bank-skewed)

typedef __attribute__((ext_vector_type(16))) __bf16 v16bf;
typedef __attribute__((ext_vector_type(8)))  float  v8f;

union V16BFCast { v16bf v; uint4 q[2]; };
union BF4Pack   { __bf16 h[4]; uint2 u; };

__device__ __forceinline__ float bf2f_raw_lo(unsigned u) { return __uint_as_float(u << 16); }
__device__ __forceinline__ float bf2f_raw_hi(unsigned u) { return __uint_as_float(u & 0xffff0000u); }

// monotonic float <-> uint key for atomic max over signed floats
__device__ __forceinline__ unsigned fkey(float f) {
    unsigned u = __float_as_uint(f);
    return (u & 0x80000000u) ? ~u : (u | 0x80000000u);
}
__device__ __forceinline__ float funkey(unsigned k) {
    unsigned u = (k & 0x80000000u) ? (k & 0x7fffffffu) : ~k;
    return __uint_as_float(u);
}

// ---------------- init: zero accumulators ----------------
__global__ void gat_init(float* __restrict__ out, float* __restrict__ denom,
                         unsigned* __restrict__ gmax) {
    size_t i = (size_t)blockIdx.x * 256 + threadIdx.x;
    if (i < (size_t)N_NODES * OUT_CH) out[i] = 0.f;
    if (i < (size_t)N_NODES * HEADS)  denom[i] = 0.f;
    if (i == 0) *gmax = 0u;   // key of most-negative value
}

// ---------------- pass A: WMMA projection + alpha epilogue ----------------
// h = X @ W^T + b ; store bf16 head-interleaved h[n][c][h] ; alpha_{dst,src}[n][h]
__global__ __launch_bounds__(256) void gat_gemm(
    const float* __restrict__ X, const float* __restrict__ W,
    const float* __restrict__ bias, const float* __restrict__ att,
    __bf16* __restrict__ Hout, float* __restrict__ alpha_dst,
    float* __restrict__ alpha_src)
{
    __shared__ __bf16 sW[128 * LDW];   // W as bf16, row j contiguous over K
    __shared__ __bf16 sH[128 * LDW];   // output tile, head-interleaved columns

    const int tid  = threadIdx.x;
    const int lane = tid & 31;
    const int wave = tid >> 5;
    const int ln   = lane & 15;
    const int hi   = lane >> 4;
    const int rowBlock = blockIdx.x * 128;

    // stage W (128x128 f32 -> bf16 in LDS), vectorized: float4 -> packed bf16 x4
    for (int j = tid; j < 128 * 32; j += 256) {
        const int r  = j >> 5;
        const int c4 = (j & 31) * 4;
        float4 f = *(const float4*)(W + r * 128 + c4);
        BF4Pack p;
        p.h[0] = (__bf16)f.x; p.h[1] = (__bf16)f.y;
        p.h[2] = (__bf16)f.z; p.h[3] = (__bf16)f.w;
        *(uint2*)&sW[r * LDW + c4] = p.u;
    }
    __syncthreads();

    // A rows for this wave (clamped; stores are guarded later)
    int arow  = rowBlock + wave * 16 + ln;
    int asafe = arow < N_NODES ? arow : (N_NODES - 1);
    const float* xrow = X + (size_t)asafe * IN_CH;

    v8f acc[8];
#pragma unroll
    for (int nt = 0; nt < 8; ++nt) acc[nt] = v8f{};

#pragma unroll
    for (int ks = 0; ks < 4; ++ks) {
        const int kb = ks * 32;
        // --- A fragment: lanes 0-15 K kb+[0..7],kb+[16..23]; lanes 16-31 +8 ---
        v16bf a;
        {
            const int k0 = kb + hi * 8;
            const int k1 = kb + 16 + hi * 8;
            float4 f0 = *(const float4*)(xrow + k0);
            float4 f1 = *(const float4*)(xrow + k0 + 4);
            float4 f2 = *(const float4*)(xrow + k1);
            float4 f3 = *(const float4*)(xrow + k1 + 4);
            a[0]=(__bf16)f0.x; a[1]=(__bf16)f0.y; a[2]=(__bf16)f0.z; a[3]=(__bf16)f0.w;
            a[4]=(__bf16)f1.x; a[5]=(__bf16)f1.y; a[6]=(__bf16)f1.z; a[7]=(__bf16)f1.w;
            a[8]=(__bf16)f2.x; a[9]=(__bf16)f2.y; a[10]=(__bf16)f2.z; a[11]=(__bf16)f2.w;
            a[12]=(__bf16)f3.x; a[13]=(__bf16)f3.y; a[14]=(__bf16)f3.z; a[15]=(__bf16)f3.w;
        }
#pragma unroll
        for (int nt = 0; nt < 8; ++nt) {
            // --- B fragment: column (nt*16+ln) = W row; lanes 0-15 K kb+[0..7],kb+[16..23],
            //     lanes 16-31 +8 (two 16B chunks per lane) ---
            const __bf16* bp = &sW[(nt * 16 + ln) * LDW + kb + hi * 8];
            V16BFCast bu;
            bu.q[0] = ((const uint4*)bp)[0];
            bu.q[1] = ((const uint4*)bp)[2];
            acc[nt] = __builtin_amdgcn_wmma_f32_16x16x32_bf16(
                false, a, false, bu.v, (short)0, acc[nt], false, false);
        }
    }

    // --- epilogue: bias add, write bf16 to sH in head-interleaved order ---
#pragma unroll
    for (int nt = 0; nt < 8; ++nt) {
        const int coln = nt * 16 + ln;                  // logical column head*32+c
        const int pidx = (coln & 31) * 4 + (coln >> 5); // interleaved: c*4+head
        const float bv = bias[coln];
#pragma unroll
        for (int v = 0; v < 8; ++v) {
            const int lrow = wave * 16 + v + 8 * hi;
            sH[lrow * LDW + pidx] = (__bf16)(acc[nt][v] + bv);
        }
    }
    __syncthreads();

    // --- coalesced store of h (64B per thread) ---
    {
        const int lrow = tid >> 1;
        const int seg  = tid & 1;
        const int grow = rowBlock + lrow;
        if (grow < N_NODES) {
            const uint4* src = (const uint4*)&sH[lrow * LDW + seg * 64];
            uint4* dst = (uint4*)(Hout + (size_t)grow * HC + seg * 64);
#pragma unroll
            for (int q = 0; q < 4; ++q) dst[q] = src[q];
        }
    }
    // --- alpha epilogue: 2 threads/row, each handles 16 channels, all 4 heads,
    //     both dst & src halves; combine across the lane pair with shfl_xor ---
    {
        const int lrow = tid >> 1;
        const int half = tid & 1;             // channels [half*16, half*16+16)
        const int grow = rowBlock + lrow;
        const __bf16* hr = &sH[lrow * LDW];
        float accd[HEADS] = {0.f, 0.f, 0.f, 0.f};
        float accs[HEADS] = {0.f, 0.f, 0.f, 0.f};
#pragma unroll
        for (int c0 = 0; c0 < 16; ++c0) {
            const int c = half * 16 + c0;
            const uint2 pv = *(const uint2*)(hr + c * 4); // heads 0..3 of channel c
            const float v0 = bf2f_raw_lo(pv.x), v1 = bf2f_raw_hi(pv.x);
            const float v2 = bf2f_raw_lo(pv.y), v3 = bf2f_raw_hi(pv.y);
            accd[0] += v0 * att[0 * 64 + c];  accs[0] += v0 * att[0 * 64 + 32 + c];
            accd[1] += v1 * att[1 * 64 + c];  accs[1] += v1 * att[1 * 64 + 32 + c];
            accd[2] += v2 * att[2 * 64 + c];  accs[2] += v2 * att[2 * 64 + 32 + c];
            accd[3] += v3 * att[3 * 64 + c];  accs[3] += v3 * att[3 * 64 + 32 + c];
        }
#pragma unroll
        for (int h = 0; h < HEADS; ++h) {
            accd[h] += __shfl_xor(accd[h], 1, 32);
            accs[h] += __shfl_xor(accs[h], 1, 32);
        }
        if (grow < N_NODES) {
            if (half == 0) {
                float4 v = make_float4(accd[0], accd[1], accd[2], accd[3]);
                ((float4*)alpha_dst)[grow] = v;
            } else {
                float4 v = make_float4(accs[0], accs[1], accs[2], accs[3]);
                ((float4*)alpha_src)[grow] = v;
            }
        }
    }
}

// ---------------- pass B1: exact global max of e[E,H] ----------------
__global__ __launch_bounds__(256) void gat_max(
    const int* __restrict__ src, const int* __restrict__ tgt,
    const float* __restrict__ ad, const float* __restrict__ as,
    unsigned* __restrict__ gmax)
{
    __shared__ unsigned red[256];
    const int e = blockIdx.x * 256 + threadIdx.x;
    float m = -3.4e38f;
    if (e < N_EDGES) {
        const int s = src[e], t = tgt[e];
        const float4 a4 = ((const float4*)ad)[t];
        const float4 s4 = ((const float4*)as)[s];
        m = fmaxf(fmaxf(a4.x + s4.x, a4.y + s4.y),
                  fmaxf(a4.z + s4.z, a4.w + s4.w));
    }
    red[threadIdx.x] = fkey(m);
    __syncthreads();
    for (int off = 128; off > 0; off >>= 1) {
        if (threadIdx.x < off) {
            unsigned o = red[threadIdx.x + off];
            if (o > red[threadIdx.x]) red[threadIdx.x] = o;
        }
        __syncthreads();
    }
    if (threadIdx.x == 0) atomicMax(gmax, red[0]);
}

// ---------------- pass B2: segment-sum denominator ----------------
__global__ __launch_bounds__(256) void gat_denom(
    const int* __restrict__ src, const int* __restrict__ tgt,
    const float* __restrict__ ad, const float* __restrict__ as,
    const unsigned* __restrict__ gmax, float* __restrict__ denom)
{
    const int e = blockIdx.x * 256 + threadIdx.x;
    if (e >= N_EDGES) return;
    const float mx = funkey(*gmax);
    const int s = src[e], t = tgt[e];
    const float4 a4 = ((const float4*)ad)[t];
    const float4 s4 = ((const float4*)as)[s];
    float ev[HEADS] = {a4.x + s4.x, a4.y + s4.y, a4.z + s4.z, a4.w + s4.w};
#pragma unroll
    for (int h = 0; h < HEADS; ++h) {
        float x = ev[h] - mx;
        x = x > 0.f ? x : 0.01f * x;        // leaky_relu
        atomicAdd(&denom[(size_t)t * HEADS + h], __expf(x));
    }
}

// ---------------- pass C: weighted scatter, head-mean fused ----------------
// one wave per edge; lane = output channel; 1 atomicAdd per (edge,channel)
__global__ __launch_bounds__(256) void gat_aggr(
    const int* __restrict__ src, const int* __restrict__ tgt,
    const float* __restrict__ ad, const float* __restrict__ as,
    const unsigned* __restrict__ gmax, const float* __restrict__ denom,
    const __bf16* __restrict__ H, float* __restrict__ out)
{
    const int lane = threadIdx.x & 31;
    const int wave = threadIdx.x >> 5;
    const size_t e = (size_t)blockIdx.x * 8 + wave;
    if (e >= N_EDGES) return;
    const int s = src[e], t = tgt[e];
    const float mx = funkey(*gmax);

    float wl = 0.f;
    if (lane < HEADS) {
        float x = ad[(size_t)t * HEADS + lane] + as[(size_t)s * HEADS + lane] - mx;
        x = x > 0.f ? x : 0.01f * x;
        wl = __expf(x) / (denom[(size_t)t * HEADS + lane] + 1e-8f);
    }
    const float w0 = __shfl(wl, 0, 32);
    const float w1 = __shfl(wl, 1, 32);
    const float w2 = __shfl(wl, 2, 32);
    const float w3 = __shfl(wl, 3, 32);

    // head-interleaved h: channel `lane`, heads 0..3 are 4 consecutive bf16 (8B)
    const uint2 pv = *(const uint2*)(H + (size_t)s * HC + lane * 4);
    const float v0 = bf2f_raw_lo(pv.x), v1 = bf2f_raw_hi(pv.x);
    const float v2 = bf2f_raw_lo(pv.y), v3 = bf2f_raw_hi(pv.y);

    const float a = 0.25f * (w0 * v0 + w1 * v1 + w2 * v2 + w3 * v3);
    atomicAdd(&out[(size_t)t * OUT_CH + lane], a);
}

// ---------------- launch ----------------
extern "C" void kernel_launch(void* const* d_in, const int* in_sizes, int n_in,
                              void* d_out, int out_size, void* d_ws, size_t ws_size,
                              hipStream_t stream) {
    const float* X    = (const float*)d_in[0];
    const int*   eidx = (const int*)d_in[1];
    const float* W    = (const float*)d_in[2];
    const float* bias = (const float*)d_in[3];
    const float* att  = (const float*)d_in[4];
    float* out = (float*)d_out;

    const int* srcI = eidx;
    const int* tgtI = eidx + N_EDGES;

    // workspace layout
    char* ws = (char*)d_ws;
    __bf16*  Hbuf = (__bf16*)ws;                       // NPAD*128 bf16 = 25.6 MB
    float*   adv  = (float*)(ws + (size_t)NPAD * HC * 2);
    float*   asv  = adv + (size_t)NPAD * HEADS;
    float*   den  = asv + (size_t)NPAD * HEADS;
    unsigned* gmx = (unsigned*)(den + (size_t)NPAD * HEADS);
    (void)ws_size; (void)in_sizes; (void)n_in; (void)out_size;

    gat_init<<<(N_NODES * OUT_CH + 255) / 256, 256, 0, stream>>>(out, den, gmx);
    gat_gemm<<<(N_NODES + 127) / 128, 256, 0, stream>>>(X, W, bias, att, Hbuf, adv, asv);
    gat_max<<<(N_EDGES + 255) / 256, 256, 0, stream>>>(srcI, tgtI, adv, asv, gmx);
    gat_denom<<<(N_EDGES + 255) / 256, 256, 0, stream>>>(srcI, tgtI, adv, asv, gmx, den);
    gat_aggr<<<(N_EDGES + 7) / 8, 256, 0, stream>>>(srcI, tgtI, adv, asv, gmx, den, Hbuf, out);
}
